// FourierLayerOriginal_26577257627813
// MI455X (gfx1250) — matrix operand order repaired and checked by reading, not compiled, not run on hardware
//
#include <hip/hip_runtime.h>
#include <hip/hip_bf16.h>

typedef __attribute__((ext_vector_type(16))) _Float16 v16h;
typedef __attribute__((ext_vector_type(8)))  float    v8f;

#define NDIM   101    // signal length / DFS width
#define DPAD   128    // K padded to 4 chunks of 32
#define KROWS  112    // output rows padded to 7 tiles of 16

// ---------------------------------------------------------------------------
// Kernel 1: fold the four constant matrices into A = (Wre*ReJ - Wim*ImJ)/N,
// split each entry into f16 hi + f16 residual, store zero-padded [112][128].
// 112*128 = 14336 entries; trivial cost, runs once per launch.
// ---------------------------------------------------------------------------
__global__ __launch_bounds__(256) void fourier_build_A(
    const float* __restrict__ Wre, const float* __restrict__ Wim,
    const float* __restrict__ ReJ, const float* __restrict__ ImJ,
    _Float16* __restrict__ Ahi, _Float16* __restrict__ Alo)
{
    int idx = blockIdx.x * blockDim.x + threadIdx.x;
    if (idx >= KROWS * DPAD) return;
    int k = idx >> 7;        // output row
    int d = idx & (DPAD - 1);

    float a = 0.0f;
    if (k < NDIM && d < NDIM) {
        float s = 0.0f;
        for (int n = 0; n < NDIM; ++n) {
            s += Wre[k * NDIM + n] * ReJ[n * NDIM + d]
               - Wim[k * NDIM + n] * ImJ[n * NDIM + d];
        }
        a = s / (float)NDIM;
    }
    _Float16 hi = (_Float16)a;
    _Float16 lo = (_Float16)(a - (float)hi);
    Ahi[idx] = hi;
    Alo[idx] = lo;
}

// ---------------------------------------------------------------------------
// Kernel 2: out[b,k] = sum_d A[k,d] * x[b,d] via v_wmma_f32_16x16x32_f16.
// One wave32 per 16 batch rows; 7 N-tiles x 4 K-chunks x 3 split-WMMAs.
// ---------------------------------------------------------------------------
__global__ __launch_bounds__(256) void fourier_gemm_wmma(
    const float*   __restrict__ x,
    const _Float16* __restrict__ Ahi,
    const _Float16* __restrict__ Alo,
    float* __restrict__ out, int Btot)
{
    const int lane = threadIdx.x & 31;       // wave32 lane
    const int wave = threadIdx.x >> 5;       // 8 waves / block
    const int m    = lane & 15;              // row-in-tile (A op) / col (B,D ops)
    const int h    = lane >> 4;              // lane half selects K sub-range

    const long b0 = (long)blockIdx.x * 128 + (long)wave * 16;

    long xr = b0 + m;
    if (xr >= Btot) xr = Btot - 1;           // clamp (stores are guarded)
    const float* __restrict__ xrow = x + xr * NDIM;

    // --- Assemble A-operand fragments (x tile), split into f16 hi/lo. ---
    // A 16x32 fragment, contiguous K order (same as B): lane(m,h), element e -> K = 32c + 16h + e
    v16h xhi[4], xlo[4];
#pragma unroll
    for (int c = 0; c < 4; ++c) {
        const int dbase = 32 * c;
#pragma unroll
        for (int e = 0; e < 16; ++e) {
            int d = dbase + 16 * h + e;
            float v = (d < NDIM) ? xrow[d] : 0.0f;
            _Float16 hi = (_Float16)v;
            xhi[c][e] = hi;
            xlo[c][e] = (_Float16)(v - (float)hi);
        }
    }

    // --- 7 output tiles of 16 columns each. ---
#pragma unroll
    for (int nt = 0; nt < 7; ++nt) {
        v8f acc = {};
        const int r = nt * 16 + m;           // B-op: column n = lane&15 -> A row
#pragma unroll
        for (int c = 0; c < 4; ++c) {
            // B 32x16 fragment, contiguous K order (same as A): lane(n,h), element e -> K = 32c + 16h + e
            const int off = r * DPAD + 32 * c + 16 * h;   // 32B-aligned
            v16h bhi = *(const v16h*)(Ahi + off);
            v16h blo = *(const v16h*)(Alo + off);
            // split-f16 accumulation: hi*hi + lo*hi + hi*lo (drop lo*lo ~2^-22)
            acc = __builtin_amdgcn_wmma_f32_16x16x32_f16(
                      false, xhi[c], false, bhi, (short)0, acc, false, false);
            acc = __builtin_amdgcn_wmma_f32_16x16x32_f16(
                      false, xlo[c], false, bhi, (short)0, acc, false, false);
            acc = __builtin_amdgcn_wmma_f32_16x16x32_f16(
                      false, xhi[c], false, blo, (short)0, acc, false, false);
        }
        // --- Store: f32 C/D layout: VGPR v, lane(h,n) -> M = 8h+v, N = n ---
        const int kout = nt * 16 + m;
#pragma unroll
        for (int v = 0; v < 8; ++v) {
            long row = b0 + 8 * h + v;
            if (row < Btot && kout < NDIM)
                out[row * NDIM + kout] = acc[v];
        }
    }
}

extern "C" void kernel_launch(void* const* d_in, const int* in_sizes, int n_in,
                              void* d_out, int out_size, void* d_ws, size_t ws_size,
                              hipStream_t stream) {
    const float* x   = (const float*)d_in[0];
    const float* Wre = (const float*)d_in[1];
    const float* Wim = (const float*)d_in[2];
    const float* ReJ = (const float*)d_in[3];
    const float* ImJ = (const float*)d_in[4];
    float* out = (float*)d_out;

    _Float16* Ahi = (_Float16*)d_ws;             // 112*128 halves = 28672 B
    _Float16* Alo = Ahi + KROWS * DPAD;          // next 28672 B (32B-aligned)

    const int Btot = in_sizes[0] / NDIM;         // 262144 for the reference

    fourier_build_A<<<(KROWS * DPAD + 255) / 256, 256, 0, stream>>>(
        Wre, Wim, ReJ, ImJ, Ahi, Alo);

    const int nblk = (Btot + 127) / 128;         // 128 batch rows per block
    fourier_gemm_wmma<<<nblk, 256, 0, stream>>>(x, Ahi, Alo, out, Btot);
}
